// AnswerScore_71201967833845
// MI455X (gfx1250) — matrix operand order repaired and checked by reading, not compiled, not run on hardware
//
#include <hip/hip_runtime.h>
#include <math.h>

#define H2 512        // 2H
#define H4 1024       // 4H
#define Bsz 128
#define Lsz 512
#define NB  384       // 3 branches * B, stacked branch-major
#define LAMDA_ 3.0f
#define EPS_   1e-8f

typedef __attribute__((ext_vector_type(16))) __bf16 v16bf;
typedef __attribute__((ext_vector_type(8)))  __bf16 v8bf;
typedef __attribute__((ext_vector_type(8)))  float  v8f;

__device__ __forceinline__ v8f wmma_bf16(v16bf a, v16bf b, v8f c) {
  // D = A(16x32 bf16) * B(32x16 bf16) + C(16x16 f32)
  return __builtin_amdgcn_wmma_f32_16x16x32_bf16(false, a, false, b, (short)0, c,
                                                 false, false);
}

// Split fp32 into bf16 hi + bf16 residual (3-term product recovers ~fp32 accuracy)
__device__ __forceinline__ void split_bf16(const float* x, v16bf& hi, v16bf& lo) {
#pragma unroll
  for (int j = 0; j < 16; ++j) {
    __bf16 h = (__bf16)x[j];
    hi[j] = h;
    lo[j] = (__bf16)(x[j] - (float)h);
  }
}

// ---------------------------------------------------------------------------
// One-time operand conversion: fp32 -> (bf16 hi, bf16 lo), optionally
// transposing (R x C) -> (C x R) so every GEMM B operand is (N,K) row-major.
// ---------------------------------------------------------------------------
__global__ void cvt_copy_kernel(const float* __restrict__ in, __bf16* __restrict__ hi,
                                __bf16* __restrict__ lo, int n) {
  const int i = blockIdx.x * 256 + threadIdx.x;
  if (i >= n) return;
  const float x = in[i];
  const __bf16 h = (__bf16)x;
  hi[i] = h;
  lo[i] = (__bf16)(x - (float)h);
}

__global__ void cvt_transpose_kernel(const float* __restrict__ in, __bf16* __restrict__ hi,
                                     __bf16* __restrict__ lo, int R, int C) {
  const int i = blockIdx.x * 256 + threadIdx.x;
  if (i >= R * C) return;
  const int r = i / C, c = i - r * C;
  const float x = in[i];
  const __bf16 h = (__bf16)x;
  hi[(size_t)c * R + r] = h;
  lo[(size_t)c * R + r] = (__bf16)(x - (float)h);
}

// ---------------------------------------------------------------------------
// C(M,N) = A(M,K) f32 * B(K,N) (+bias, optional relu); one wave per 16x16 tile.
// B supplied pre-split as bf16 hi/lo in (N,K) row-major (ldb=K): B fragment is
// two contiguous 32B v16bf loads per lane. A split on the fly (A is small).
// Fragment layouts used here (wave32; A and B share this contiguous k order):
//   A 16x32 bf16 : lane m=lane&15; halves j: k = k0 + (lane>=16?16:0) + j
//   B 32x16 bf16 : lane n=lane&15; halves j: k = k0 + (lane>=16?16:0) + j
//   C 16x16 f32  : vgpr r: row = r + (lane>=16?8:0), col = lane&15
// ---------------------------------------------------------------------------
__global__ void gemm_wmma_kernel(const float* __restrict__ A,
                                 const __bf16* __restrict__ Bhi, const __bf16* __restrict__ Blo,
                                 const float* __restrict__ bias, float* __restrict__ C,
                                 int K, int lda, int ldb, int ldc, int relu) {
  const int lane = threadIdx.x & 31;
  const int hv   = lane >> 4;
  const int lm   = lane & 15;
  const int n0   = blockIdx.x * 16;
  const int row0 = blockIdx.y * 16;
  v8f acc = {};
  for (int k0 = 0; k0 < K; k0 += 32) {
    float af[16];
    const float* ar = A + (size_t)(row0 + lm) * lda + k0 + hv * 16;
#pragma unroll
    for (int j = 0; j < 8; ++j) { af[j] = ar[j]; af[8 + j] = ar[8 + j]; }
    v16bf ah, al;
    split_bf16(af, ah, al);
    const size_t boff = (size_t)(n0 + lm) * ldb + k0 + hv * 16;
    const v16bf bh = *(const v16bf*)&Bhi[boff];
    const v16bf bl = *(const v16bf*)&Blo[boff];
    acc = wmma_bf16(ah, bh, acc);
    acc = wmma_bf16(al, bh, acc);
    acc = wmma_bf16(ah, bl, acc);
  }
  const float bv = bias ? bias[n0 + lm] : 0.f;
#pragma unroll
  for (int r = 0; r < 8; ++r) {
    float v = acc[r] + bv;
    if (relu) v = fmaxf(v, 0.f);
    C[(size_t)(row0 + r + hv * 8) * ldc + (n0 + lm)] = v;
  }
}

// ---------------------------------------------------------------------------
// m_norm[row] = max(|| m[row,:] @ w2 + b2 ||, EPS) without materializing mtt.
// 128 threads (4 waves) per 16-row block: the A tile is converted to bf16
// hi/lo in LDS exactly once, then each wave sweeps 8 of the 32 N-tiles
// (ds_load_b128 fragments + global v16bf weight fragments + 3 WMMAs per chunk).
// Row sums-of-squares are combined across waves with LDS float atomics.
// ---------------------------------------------------------------------------
__global__ void mnorm_kernel(const float* __restrict__ m,
                             const __bf16* __restrict__ w2hi, const __bf16* __restrict__ w2lo,
                             const float* __restrict__ b2, float* __restrict__ mnorm) {
  __shared__ __bf16 Ahi_s[16 * H2];
  __shared__ __bf16 Alo_s[16 * H2];
  __shared__ float ssq[16];
  const int tid  = threadIdx.x;          // 0..127
  const int row0 = blockIdx.x * 16;
  if (tid < 16) ssq[tid] = 0.f;
  for (int t = tid; t < 16 * H2; t += 128) {
    const int r = t >> 9, k = t & (H2 - 1);
    const float x = m[(size_t)(row0 + r) * H2 + k];
    const __bf16 h = (__bf16)x;
    Ahi_s[t] = h;
    Alo_s[t] = (__bf16)(x - (float)h);
  }
  __syncthreads();
  const int wv = tid >> 5, lane = tid & 31;
  const int hv = lane >> 4, lm = lane & 15;
  float sumsq[8];
#pragma unroll
  for (int r = 0; r < 8; ++r) sumsq[r] = 0.f;
  for (int i = 0; i < 8; ++i) {
    const int n0 = (wv + 4 * i) * 16;
    v8f acc = {};
    for (int k0 = 0; k0 < H2; k0 += 32) {
      const int abase = lm * H2 + k0 + hv * 16;
      const v8bf a0 = *(const v8bf*)&Ahi_s[abase];
      const v8bf a1 = *(const v8bf*)&Ahi_s[abase + 8];
      const v8bf l0 = *(const v8bf*)&Alo_s[abase];
      const v8bf l1 = *(const v8bf*)&Alo_s[abase + 8];
      v16bf ah, al;
#pragma unroll
      for (int j = 0; j < 8; ++j) {
        ah[j] = a0[j]; ah[8 + j] = a1[j];
        al[j] = l0[j]; al[8 + j] = l1[j];
      }
      const size_t boff = (size_t)(n0 + lm) * H2 + k0 + hv * 16;
      const v16bf bh = *(const v16bf*)&w2hi[boff];
      const v16bf bl = *(const v16bf*)&w2lo[boff];
      acc = wmma_bf16(ah, bh, acc);
      acc = wmma_bf16(al, bh, acc);
      acc = wmma_bf16(ah, bl, acc);
    }
    const float bv = b2[n0 + lm];
#pragma unroll
    for (int r = 0; r < 8; ++r) { const float v = acc[r] + bv; sumsq[r] += v * v; }
  }
#pragma unroll
  for (int r = 0; r < 8; ++r) {
    float s = sumsq[r];
#pragma unroll
    for (int off = 1; off < 16; off <<= 1) s += __shfl_xor(s, off, 32);
    if (lm == 0) atomicAdd(&ssq[r + hv * 8], s);
  }
  __syncthreads();
  if (tid < 16) mnorm[row0 + tid] = fmaxf(sqrtf(ssq[tid]), EPS_);
}

__global__ void rownorm_kernel(const float* __restrict__ X, float* __restrict__ nrm,
                               int ncols) {
  const int i = blockIdx.x, lane = threadIdx.x;
  float s = 0.f;
  for (int d = lane; d < ncols; d += 32) { const float v = X[(size_t)i * ncols + d]; s += v * v; }
#pragma unroll
  for (int off = 16; off > 0; off >>= 1) s += __shfl_xor(s, off, 32);
  if (lane == 0) nrm[i] = fmaxf(sqrtf(s), EPS_);
}

// ---------------------------------------------------------------------------
// Fused attention per (branch,batch) row i: dot = m[b] @ q[i] (w2 folded into
// q), mask+scale, softmax over L, f_att = p @ m[b]. Two streaming passes over
// the 1 MB m[b] slice (L2-resident after hop 0). 256 threads = 8 waves.
// ---------------------------------------------------------------------------
__global__ void attention_kernel(const float* __restrict__ m, const int* __restrict__ mask,
                                 const float* __restrict__ mnorm, const float* __restrict__ q,
                                 const float* __restrict__ snorm, float* __restrict__ fatt) {
  const int i = blockIdx.x;          // 0..383  (branch-major)
  const int b = i & (Bsz - 1);       // shared m slice across branches
  const float* mb = m + (size_t)b * Lsz * H2;
  const float* qi = q + (size_t)i * H2;
  __shared__ float att[Lsz];
  __shared__ float red[8];
  const int tid = threadIdx.x;
  const int wv = tid >> 5, lane = tid & 31;
  const float sn = snorm[i];
  for (int l = wv; l < Lsz; l += 8) {
    float p = 0.f;
    const float* mr = mb + (size_t)l * H2;
    for (int d = lane; d < H2; d += 32) p += mr[d] * qi[d];
#pragma unroll
    for (int off = 16; off > 0; off >>= 1) p += __shfl_xor(p, off, 32);
    if (lane == 0) {
      float a;
      if (mask[b * Lsz + l] == 0) a = -__builtin_inff();
      else a = LAMDA_ * p / (mnorm[(size_t)b * Lsz + l] * sn);
      att[l] = a;
    }
  }
  __syncthreads();
  float mx = -__builtin_inff();
  for (int l = tid; l < Lsz; l += 256) mx = fmaxf(mx, att[l]);
#pragma unroll
  for (int off = 16; off > 0; off >>= 1) mx = fmaxf(mx, __shfl_xor(mx, off, 32));
  if (lane == 0) red[wv] = mx;
  __syncthreads();
  mx = red[0];
#pragma unroll
  for (int w = 1; w < 8; ++w) mx = fmaxf(mx, red[w]);
  __syncthreads();                          // done reading red before reuse
  float sum = 0.f;
  for (int l = tid; l < Lsz; l += 256) {
    const float e = __expf(att[l] - mx);
    att[l] = e;
    sum += e;
  }
#pragma unroll
  for (int off = 16; off > 0; off >>= 1) sum += __shfl_xor(sum, off, 32);
  if (lane == 0) red[wv] = sum;
  __syncthreads();
  float tot = 0.f;
#pragma unroll
  for (int w = 0; w < 8; ++w) tot += red[w];
  const float inv = 1.f / tot;
  float a0 = 0.f, a1 = 0.f;
  for (int l = 0; l < Lsz; ++l) {
    const float pw = att[l] * inv;
    a0 += pw * mb[(size_t)l * H2 + tid];
    a1 += pw * mb[(size_t)l * H2 + tid + 256];
  }
  fatt[(size_t)i * H2 + tid]       = a0;
  fatt[(size_t)i * H2 + tid + 256] = a1;
}

// GRU gate math (torch order r,z,n); updates st in place.
__global__ void gru_gate_kernel(const float* __restrict__ gx, const float* __restrict__ gh,
                                float* __restrict__ st) {
  const int idx = blockIdx.x * blockDim.x + threadIdx.x;
  if (idx >= NB * H4) return;
  const int i = idx >> 10;          // /H4
  const int j = idx & (H4 - 1);
  const size_t base = (size_t)i * 3 * H4 + j;
  float r  = gx[base]      + gh[base];
  float z  = gx[base + H4] + gh[base + H4];
  const float xn = gx[base + 2 * H4];
  const float hn = gh[base + 2 * H4];
  r = 1.f / (1.f + __expf(-r));
  z = 1.f / (1.f + __expf(-z));
  const float n = tanhf(xn + r * hn);
  const float h = st[idx];
  st[idx] = (1.f - z) * n + z * h;
}

__global__ void head_kernel(const float* __restrict__ h, const float* __restrict__ w5,
                            const float* __restrict__ b5, float* __restrict__ res) {
  const int i = blockIdx.x, lane = threadIdx.x;
  float s = 0.f;
  for (int d = lane; d < H4; d += 32) s += h[(size_t)i * H4 + d] * w5[d];
#pragma unroll
  for (int off = 16; off > 0; off >>= 1) s += __shfl_xor(s, off, 32);
  if (lane == 0) res[i] = s + b5[0];
}

__global__ void logsoftmax_kernel(const float* __restrict__ res, float* __restrict__ out) {
  const int b = threadIdx.x;        // 128 threads
  const float a0 = res[b], a1 = res[Bsz + b], a2 = res[2 * Bsz + b];
  const float mx  = fmaxf(a0, fmaxf(a1, a2));
  const float lse = mx + logf(__expf(a0 - mx) + __expf(a1 - mx) + __expf(a2 - mx));
  out[b * 3 + 0] = a0 - lse;
  out[b * 3 + 1] = a1 - lse;
  out[b * 3 + 2] = a2 - lse;
}

extern "C" void kernel_launch(void* const* d_in, const int* in_sizes, int n_in,
                              void* d_out, int out_size, void* d_ws, size_t ws_size,
                              hipStream_t stream) {
  (void)in_sizes; (void)n_in; (void)out_size; (void)ws_size;
  const float* m     = (const float*)d_in[0];
  const float* s1    = (const float*)d_in[1];
  const float* s2    = (const float*)d_in[2];
  const float* s3    = (const float*)d_in[3];
  const int*   mmask = (const int*)  d_in[4];
  const float* w2w   = (const float*)d_in[5];
  const float* w2b   = (const float*)d_in[6];
  const float* w3w   = (const float*)d_in[7];
  const float* w3b   = (const float*)d_in[8];
  const float* w4w   = (const float*)d_in[9];
  const float* w4b   = (const float*)d_in[10];
  const float* w5w   = (const float*)d_in[11];
  const float* w5b   = (const float*)d_in[12];
  const float* gwi   = (const float*)d_in[13];
  const float* gwh   = (const float*)d_in[14];
  const float* gbi   = (const float*)d_in[15];
  const float* gbh   = (const float*)d_in[16];
  float* out = (float*)d_out;

  float* ws = (float*)d_ws;
  float* mnorm = ws; ws += Bsz * Lsz;        // 65536
  float* S     = ws; ws += NB * H4;          // stacked states, branch-major
  float* stt   = ws; ws += NB * H2;
  float* snorm = ws; ws += NB;
  float* q     = ws; ws += NB * H2;
  float* fatt  = ws; ws += NB * H2;
  float* gx    = ws; ws += NB * 3 * H4;
  float* gh    = ws; ws += NB * 3 * H4;
  float* hbuf  = ws; ws += NB * H4;
  float* res   = ws; ws += NB;

  // bf16 hi/lo weight copies, all in (N,K) row-major so B fragments are
  // contiguous 32B loads.  (counts all multiples of 16 -> 32B-aligned slices)
  __bf16* bp = (__bf16*)ws;
  __bf16* w2nh = bp; bp += H2 * H2;   // transpose(w2)  : mnorm B
  __bf16* w2nl = bp; bp += H2 * H2;
  __bf16* w2th = bp; bp += H2 * H2;   // w2 as-is       : q = stt @ w2^T
  __bf16* w2tl = bp; bp += H2 * H2;
  __bf16* w3nh = bp; bp += H4 * H2;   // transpose(w3)
  __bf16* w3nl = bp; bp += H4 * H2;
  __bf16* w4nh = bp; bp += H4 * H4;   // transpose(w4)
  __bf16* w4nl = bp; bp += H4 * H4;
  __bf16* gwih = bp; bp += 3 * H4 * H2;  // gru_wi as-is (already (N,K))
  __bf16* gwil = bp; bp += 3 * H4 * H2;
  __bf16* gwhh = bp; bp += 3 * H4 * H4;  // gru_wh as-is
  __bf16* gwhl = bp; bp += 3 * H4 * H4;

  // Stack the three branch states: rows [0,128)=s1, [128,256)=s2, [256,384)=s3
  hipMemcpyAsync(S,                s1, sizeof(float) * Bsz * H4, hipMemcpyDeviceToDevice, stream);
  hipMemcpyAsync(S + Bsz * H4,     s2, sizeof(float) * Bsz * H4, hipMemcpyDeviceToDevice, stream);
  hipMemcpyAsync(S + 2 * Bsz * H4, s3, sizeof(float) * Bsz * H4, hipMemcpyDeviceToDevice, stream);

  // One-time weight conversion to pre-split bf16 (N,K)
  cvt_transpose_kernel<<<(H2 * H2 + 255) / 256, 256, 0, stream>>>(w2w, w2nh, w2nl, H2, H2);
  cvt_copy_kernel<<<(H2 * H2 + 255) / 256, 256, 0, stream>>>(w2w, w2th, w2tl, H2 * H2);
  cvt_transpose_kernel<<<(H4 * H2 + 255) / 256, 256, 0, stream>>>(w3w, w3nh, w3nl, H4, H2);
  cvt_transpose_kernel<<<(H4 * H4 + 255) / 256, 256, 0, stream>>>(w4w, w4nh, w4nl, H4, H4);
  cvt_copy_kernel<<<(3 * H4 * H2 + 255) / 256, 256, 0, stream>>>(gwi, gwih, gwil, 3 * H4 * H2);
  cvt_copy_kernel<<<(3 * H4 * H4 + 255) / 256, 256, 0, stream>>>(gwh, gwhh, gwhl, 3 * H4 * H4);

  // Loop-invariant ||m @ w2 + b2|| row norms (mtt never materialized)
  mnorm_kernel<<<dim3((Bsz * Lsz) / 16), 128, 0, stream>>>(m, w2nh, w2nl, w2b, mnorm);

  for (int hop = 0; hop < 3; ++hop) {
    // stt = S @ w3 + b3             (384 x 512, K=1024)
    gemm_wmma_kernel<<<dim3(H2 / 16, NB / 16), 32, 0, stream>>>(
        S, w3nh, w3nl, w3b, stt, H4, H4, H4, H2, 0);
    rownorm_kernel<<<NB, 32, 0, stream>>>(stt, snorm, H2);
    // q = stt @ w2^T  (folds w2 into the query so attention reads m, not mtt)
    gemm_wmma_kernel<<<dim3(H2 / 16, NB / 16), 32, 0, stream>>>(
        stt, w2th, w2tl, nullptr, q, H2, H2, H2, H2, 0);
    attention_kernel<<<NB, 256, 0, stream>>>(m, mmask, mnorm, q, snorm, fatt);
    // gx = fatt @ gru_wi^T + bi     (384 x 3072, K=512)
    gemm_wmma_kernel<<<dim3(3 * H4 / 16, NB / 16), 32, 0, stream>>>(
        fatt, gwih, gwil, gbi, gx, H2, H2, H2, 3 * H4, 0);
    // gh = S @ gru_wh^T + bh        (384 x 3072, K=1024)
    gemm_wmma_kernel<<<dim3(3 * H4 / 16, NB / 16), 32, 0, stream>>>(
        S, gwhh, gwhl, gbh, gh, H4, H4, H4, 3 * H4, 0);
    gru_gate_kernel<<<(NB * H4 + 255) / 256, 256, 0, stream>>>(gx, gh, S);
  }

  // h = relu(S @ w4 + b4)           (384 x 1024, K=1024)
  gemm_wmma_kernel<<<dim3(H4 / 16, NB / 16), 32, 0, stream>>>(
      S, w4nh, w4nl, w4b, hbuf, H4, H4, H4, H4, 1);
  head_kernel<<<NB, 32, 0, stream>>>(hbuf, w5w, w5b, res);
  logsoftmax_kernel<<<1, Bsz, 0, stream>>>(res, out);
}
